// LocalGNO_61658550501614
// MI455X (gfx1250) — matrix operand order repaired and checked by reading, not compiled, not run on hardware
//
#include <hip/hip_runtime.h>

#define BB   8
#define NN   16384
#define HID  96
#define KNB  4
#define NL   2
#define TILE 64
#define HALO 4
#define MROW 80          // padded halo rows (TILE + 2*HALO = 72 -> 80)
#define LN_EPS 1e-5f

typedef __attribute__((ext_vector_type(16))) __bf16 v16bf;
typedef __attribute__((ext_vector_type(8)))  float  v8f;
typedef __attribute__((ext_vector_type(4)))  float  f32x4;
typedef unsigned int u32x4 __attribute__((ext_vector_type(4)));
typedef unsigned int u32x8 __attribute__((ext_vector_type(8)));

// ---------------------------------------------------------------------------
// Tensor Data Mover: async 2D tile (nrows x row_bytes, row pitch stride_bytes)
// global -> LDS. D# per CDNA5 ISA. 2-D form (VADDR2/3 = NULL).
// Issue from ONE wave only; completion via s_wait_tensorcnt + barrier.
// ---------------------------------------------------------------------------
__device__ inline void tdm_load_2d(unsigned lds_addr, const void* gsrc,
                                   unsigned row_bytes, unsigned nrows,
                                   unsigned stride_bytes) {
  unsigned long long ga = (unsigned long long)(uintptr_t)gsrc;
  u32x4 g0;
  g0[0] = 1u;                                    // count=1, user mode
  g0[1] = lds_addr;                              // LDS byte address
  g0[2] = (unsigned)ga;                          // global_addr[31:0]
  g0[3] = (unsigned)(ga >> 32) | (2u << 30);     // global_addr[56:32] | type=2
  u32x8 g1;
  g1[0] = 0u;                                    // wg_mask=0, data_size=1B
  g1[1] = (row_bytes & 0xFFFFu) << 16;           // tensor_dim0[15:0] @bit48
  g1[2] = (row_bytes >> 16) & 0xFFFFu;           // tensor_dim0[31:16]; TD1 lo16=0
  g1[3] = 0x10u | (row_bytes << 16);             // TD1 hi16 (=1M rows) | tile_dim0
  g1[4] = nrows & 0xFFFFu;                       // tile_dim1 (tile_dim2=0)
  g1[5] = stride_bytes;                          // tensor_dim0_stride[31:0]
  g1[6] = 0u;                                    // stride hi / dim1_stride lo
  g1[7] = 0u;
  asm volatile("tensor_load_to_lds %0, %1" :: "s"(g0), "s"(g1) : "memory");
}

// ---------------------------------------------------------------------------
// WMMA fragment helpers (ISA 7.12.2 layouts, wave32)
// ---------------------------------------------------------------------------
__device__ inline v16bf load_a_frag(const __bf16* A, int lda, int m0, int k0) {
  const int lane = threadIdx.x & 31;
  const int g = lane >> 4;
  const __bf16* p = A + (size_t)(m0 + (lane & 15)) * lda + k0 + 8 * g;
  v16bf r;
  ((f32x4*)&r)[0] = *(const f32x4*)(p);
  ((f32x4*)&r)[1] = *(const f32x4*)(p + 16);
  return r;
}

__device__ inline v16bf load_b_frag(const __bf16* BT, int ldb, int n0, int k0) {
  const int lane = threadIdx.x & 31;
  const int g = lane >> 4;
  const __bf16* p = BT + (size_t)(n0 + (lane & 15)) * ldb + k0 + 8 * g;
  v16bf r;
  ((f32x4*)&r)[0] = *(const f32x4*)(p);
  ((f32x4*)&r)[1] = *(const f32x4*)(p + 16);
  return r;
}

__device__ inline v8f wmma_acc(const __bf16* A, int lda, const __bf16* BT, int ldb,
                               int m0, int n0, int Kdim, float binit) {
  v8f acc;
#pragma unroll
  for (int v = 0; v < 8; ++v) acc[v] = binit;
  for (int k0 = 0; k0 < Kdim; k0 += 32) {
    v16bf a = load_a_frag(A, lda, m0, k0);
    v16bf b = load_b_frag(BT, ldb, n0, k0);
    acc = __builtin_amdgcn_wmma_f32_16x16x32_bf16(false, a, false, b, (short)0, acc,
                                                  false, false);
  }
  return acc;
}

__device__ inline float siluf(float v) { return v / (1.0f + __expf(-v)); }

// ---------------------------------------------------------------------------
// Weight prep: f32 (KxNn row-major) -> bf16 W^T layout dst[n*K + k]
// ---------------------------------------------------------------------------
__global__ __launch_bounds__(256) void transpose_bf16(const float* __restrict__ src,
                                                      __bf16* __restrict__ dst,
                                                      int Kd, int Nn) {
  int idx = blockIdx.x * 256 + threadIdx.x;
  if (idx < Kd * Nn) {
    int n = idx / Kd, k = idx % Kd;
    dst[idx] = (__bf16)src[(size_t)k * Nn + n];
  }
}

// msg_w1 (193x96): combined B^T (n<192, k<96) for [S|T] = h @ [W1a|W1b]
__global__ __launch_bounds__(256) void prep_msg1(const float* __restrict__ w,
                                                 __bf16* __restrict__ dst) {
  int idx = blockIdx.x * 256 + threadIdx.x;
  if (idx < 192 * 96) {
    int n = idx / 96, k = idx % 96;
    float v = (n < 96) ? w[(size_t)k * 96 + n] : w[(size_t)(96 + k) * 96 + (n - 96)];
    dst[(size_t)n * 96 + k] = (__bf16)v;
  }
}

// ---------------------------------------------------------------------------
// Embed: h0 = MLP(x, 4->96 silu 96->96); writes f32 h + bf16 mirror
// ---------------------------------------------------------------------------
__global__ __launch_bounds__(256) void embed_kernel(
    const float* __restrict__ x, const float* __restrict__ w1,
    const float* __restrict__ b1, const __bf16* __restrict__ wt2,
    const float* __restrict__ b2, float* __restrict__ h0,
    __bf16* __restrict__ hm0) {
  __shared__ float  xs[128 * 4];
  __shared__ __bf16 t1[128 * 96];
  const int tid = threadIdx.x;
  const int b = blockIdx.y;
  const size_t base = (size_t)b * NN + (size_t)blockIdx.x * 128;

  for (int i = tid; i < 128 * 4; i += 256) xs[i] = x[base * 4 + i];
  __syncthreads();

  for (int idx = tid; idx < 128 * 96; idx += 256) {
    int n = idx / 96, c = idx % 96;
    float acc = b1[c];
#pragma unroll
    for (int k = 0; k < 4; ++k) acc += xs[n * 4 + k] * w1[k * 96 + c];
    t1[idx] = (__bf16)siluf(acc);
  }
  __syncthreads();

  const int wave = tid >> 5, lane = tid & 31, g = lane >> 4;
  for (int job = wave; job < 8 * 6; job += 8) {
    int m0 = (job / 6) * 16, n0 = (job % 6) * 16;
    v8f acc = wmma_acc(t1, 96, wt2, 96, m0, n0, 96, b2[n0 + (lane & 15)]);
#pragma unroll
    for (int v = 0; v < 8; ++v) {
      size_t o = (base + m0 + v + 8 * g) * 96 + n0 + (lane & 15);
      h0[o] = acc[v];
      hm0[o] = (__bf16)acc[v];
    }
  }
}

// ---------------------------------------------------------------------------
// One GNO layer over a 64-node tile (+4 halo). 256 threads, TDM-staged A tile.
// ---------------------------------------------------------------------------
__global__ __launch_bounds__(256) void layer_kernel(
    const float* __restrict__ h_in, const __bf16* __restrict__ hm_in,
    float* __restrict__ h_out, __bf16* __restrict__ hm_out,
    const float* __restrict__ coord,
    const float* __restrict__ msg_w1f,   // f32 193x96 (dc row 192)
    const __bf16* __restrict__ wt_msg1,  // 192x96
    const float* __restrict__ msg_b1,
    const __bf16* __restrict__ wt_msg2,  // 96x96
    const float* __restrict__ msg_b2,
    const __bf16* __restrict__ wt_upd1,  // 96x192
    const float* __restrict__ upd_b1,
    const __bf16* __restrict__ wt_upd2,  // 96x96
    const float* __restrict__ upd_b2,
    const float* __restrict__ lng, const float* __restrict__ lnb) {
  extern __shared__ char smem[];
  float*  coord_s = (float*)(smem);                 //   320 B (80 f32)
  __bf16* hb      = (__bf16*)(smem + 320);          // 80x96 bf16 = 15360
  float*  ST      = (float*)(smem + 15680);         // 80x192 f32 = 61440
  __bf16* U       = (__bf16*)(smem + 77120);        // 64x96 bf16 = 12288
  __bf16* aggb    = (__bf16*)(smem + 89408);        // 64x96 bf16 = 12288
  __bf16* midb    = (__bf16*)(smem + 101696);       // 64x96 bf16 = 12288
  float*  dh      = (float*)(smem + 113984);        // 64x96 f32  = 24576

  const int tid = threadIdx.x;
  const int wave = tid >> 5, lane = tid & 31, g = lane >> 4;
  const int b = blockIdx.y;
  const int tstart = blockIdx.x * TILE;
  const size_t rowbase = (size_t)b * NN;

  if (tid == 0) __builtin_prefetch(h_in + (rowbase + tstart) * 96, 0, 0);

  // valid halo rows r in [r0, r1): gi = tstart - 4 + r in [0, NN)
  const int r0 = (tstart == 0) ? HALO : 0;
  const int r1 = min(TILE + 2 * HALO, NN + HALO - tstart);

  // ---- TDM: DMA bf16 h rows [r0,r1) into LDS (wave 0 only; EXEC-independent)
  if (wave == 0) {
    tdm_load_2d((unsigned)(uintptr_t)hb + (unsigned)r0 * 192,
                hm_in + (rowbase + tstart - HALO + r0) * 96,
                192u, (unsigned)(r1 - r0), 192u);
  }
  // ---- zero pad/invalid rows; stage coords (other waves do this meanwhile)
  for (int r = tid; r < MROW; r += 256) {
    int gi = tstart - HALO + r;
    coord_s[r] = (gi >= 0 && gi < NN) ? coord[rowbase + gi] : 0.0f;
  }
  for (int idx = tid; idx < MROW * 96; idx += 256) {
    int r = idx / 96;
    if (r < r0 || r >= r1) hb[idx] = (__bf16)0.0f;
  }
  if (wave == 0) __builtin_amdgcn_s_wait_tensorcnt(0);
  __syncthreads();

  // ---- [S|T] = h @ [W1a|W1b] (+b1 on S half): M=80, N=192, K=96 ----
  for (int job = wave; job < 5 * 12; job += 8) {
    int m0 = (job / 12) * 16, n0 = (job % 12) * 16;
    int nc = n0 + (lane & 15);
    float bi = (nc < 96) ? msg_b1[nc] : 0.0f;
    v8f acc = wmma_acc(hb, 96, wt_msg1, 96, m0, n0, 96, bi);
#pragma unroll
    for (int v = 0; v < 8; ++v) ST[(m0 + v + 8 * g) * 192 + nc] = acc[v];
  }
  __syncthreads();

  // ---- U[i] = sum_j silu(S[i] + T[j] + (c_j - c_i) * w1c)  (VALU phase) ----
  for (int idx = tid; idx < TILE * 96; idx += 256) {
    int i = idx / 96, c = idx % 96;
    int ri = i + HALO;
    int gi = tstart + i;
    float S   = ST[ri * 192 + c];
    float w1c = msg_w1f[192 * 96 + c];
    float ci  = coord_s[ri];
    float acc = 0.0f;
#pragma unroll
    for (int off = 1; off <= KNB; ++off) {
      if (gi + off < NN) {
        float pre = S + ST[(ri + off) * 192 + 96 + c] + (coord_s[ri + off] - ci) * w1c;
        acc += siluf(pre);
      }
      if (gi - off >= 0) {
        float pre = S + ST[(ri - off) * 192 + 96 + c] + (coord_s[ri - off] - ci) * w1c;
        acc += siluf(pre);
      }
    }
    U[idx] = (__bf16)acc;
  }
  __syncthreads();

  // ---- agg = U @ msg_w2 / count + msg_b2 : M=64, N=96, K=96 ----
  for (int job = wave; job < 4 * 6; job += 8) {
    int m0 = (job / 6) * 16, n0 = (job % 6) * 16;
    float b2n = msg_b2[n0 + (lane & 15)];
    v8f acc = wmma_acc(U, 96, wt_msg2, 96, m0, n0, 96, 0.0f);
#pragma unroll
    for (int v = 0; v < 8; ++v) {
      int m = m0 + v + 8 * g;
      int gi = tstart + m;
      float cnt = (float)(min(gi, KNB) + min(NN - 1 - gi, KNB));
      cnt = fmaxf(cnt, 1.0f);
      aggb[m * 96 + n0 + (lane & 15)] = (__bf16)(acc[v] / cnt + b2n);
    }
  }
  __syncthreads();

  // ---- update MLP stage 1: silu([h|agg] @ upd_w1 + b1), split-K 96+96 ----
  for (int job = wave; job < 4 * 6; job += 8) {
    int m0 = (job / 6) * 16, n0 = (job % 6) * 16;
    v8f acc;
#pragma unroll
    for (int v = 0; v < 8; ++v) acc[v] = upd_b1[n0 + (lane & 15)];
    for (int k0 = 0; k0 < 96; k0 += 32) {
      v16bf a = load_a_frag(hb, 96, m0 + HALO, k0);
      v16bf bm = load_b_frag(wt_upd1, 192, n0, k0);
      acc = __builtin_amdgcn_wmma_f32_16x16x32_bf16(false, a, false, bm, (short)0, acc,
                                                    false, false);
    }
    for (int k0 = 96; k0 < 192; k0 += 32) {
      v16bf a = load_a_frag(aggb, 96, m0, k0 - 96);
      v16bf bm = load_b_frag(wt_upd1, 192, n0, k0);
      acc = __builtin_amdgcn_wmma_f32_16x16x32_bf16(false, a, false, bm, (short)0, acc,
                                                    false, false);
    }
#pragma unroll
    for (int v = 0; v < 8; ++v)
      midb[(m0 + v + 8 * g) * 96 + n0 + (lane & 15)] = (__bf16)siluf(acc[v]);
  }
  __syncthreads();

  // ---- update MLP stage 2: dh = mid @ upd_w2 + b2 ----
  for (int job = wave; job < 4 * 6; job += 8) {
    int m0 = (job / 6) * 16, n0 = (job % 6) * 16;
    v8f acc = wmma_acc(midb, 96, wt_upd2, 96, m0, n0, 96, upd_b2[n0 + (lane & 15)]);
#pragma unroll
    for (int v = 0; v < 8; ++v) dh[(m0 + v + 8 * g) * 96 + n0 + (lane & 15)] = acc[v];
  }
  __syncthreads();

  // ---- residual + LayerNorm; write f32 h_out + bf16 mirror ----
  if (tid < TILE) {
    int gi = tstart + tid;
    const float* hrow = h_in + (rowbase + gi) * 96;
    float* drow = dh + tid * 96;
    float mu = 0.0f;
    for (int c = 0; c < 96; ++c) {
      float y = hrow[c] + drow[c];
      drow[c] = y;
      mu += y;
    }
    mu *= (1.0f / 96.0f);
    float var = 0.0f;
    for (int c = 0; c < 96; ++c) {
      float d = drow[c] - mu;
      var += d * d;
    }
    var *= (1.0f / 96.0f);
    float rs = __frsqrt_rn(var + LN_EPS);
    float* of = h_out + (rowbase + gi) * 96;
    __bf16* om = hm_out + (rowbase + gi) * 96;
    for (int c = 0; c < 96; ++c) {
      float y = (drow[c] - mu) * rs * lng[c] + lnb[c];
      of[c] = y;
      om[c] = (__bf16)y;
    }
  }
}

// ---------------------------------------------------------------------------
// Decode: out = silu(h @ dec_w1 + b1) @ dec_w2 + b2. TDM-staged 128-node tiles
// ---------------------------------------------------------------------------
__global__ __launch_bounds__(256) void decode_kernel(
    const __bf16* __restrict__ hm, const __bf16* __restrict__ wt1,
    const float* __restrict__ b1, const float* __restrict__ w2,
    const float* __restrict__ b2, float* __restrict__ out) {
  __shared__ __bf16 hbL[128 * 96];
  __shared__ __bf16 midL[128 * 96];
  const int tid = threadIdx.x;
  const int wave = tid >> 5, lane = tid & 31, g = lane >> 4;
  const int b = blockIdx.y;
  const size_t base = (size_t)b * NN + (size_t)blockIdx.x * 128;

  if (wave == 0) {
    tdm_load_2d((unsigned)(uintptr_t)hbL, hm + base * 96, 192u, 128u, 192u);
    __builtin_amdgcn_s_wait_tensorcnt(0);
  }
  __syncthreads();

  for (int job = wave; job < 8 * 6; job += 8) {
    int m0 = (job / 6) * 16, n0 = (job % 6) * 16;
    v8f acc = wmma_acc(hbL, 96, wt1, 96, m0, n0, 96, b1[n0 + (lane & 15)]);
#pragma unroll
    for (int v = 0; v < 8; ++v)
      midL[(m0 + v + 8 * g) * 96 + n0 + (lane & 15)] = (__bf16)siluf(acc[v]);
  }
  __syncthreads();

  if (tid < 128) {
    float acc = b2[0];
    for (int c = 0; c < 96; ++c) acc += (float)midL[tid * 96 + c] * w2[c];
    out[base + tid] = acc;
  }
}

// ---------------------------------------------------------------------------
extern "C" void kernel_launch(void* const* d_in, const int* in_sizes, int n_in,
                              void* d_out, int out_size, void* d_ws, size_t ws_size,
                              hipStream_t stream) {
  const float* x        = (const float*)d_in[0];
  const float* coord    = (const float*)d_in[1];
  const float* embed_w1 = (const float*)d_in[2];
  const float* embed_b1 = (const float*)d_in[3];
  const float* embed_w2 = (const float*)d_in[4];
  const float* embed_b2 = (const float*)d_in[5];
  const float* msg_w1   = (const float*)d_in[6];
  const float* msg_b1   = (const float*)d_in[7];
  const float* msg_w2   = (const float*)d_in[8];
  const float* msg_b2   = (const float*)d_in[9];
  const float* upd_w1   = (const float*)d_in[10];
  const float* upd_b1   = (const float*)d_in[11];
  const float* upd_w2   = (const float*)d_in[12];
  const float* upd_b2   = (const float*)d_in[13];
  const float* ln_g     = (const float*)d_in[14];
  const float* ln_b     = (const float*)d_in[15];
  const float* dec_w1   = (const float*)d_in[16];
  const float* dec_b1   = (const float*)d_in[17];
  const float* dec_w2   = (const float*)d_in[18];
  const float* dec_b2   = (const float*)d_in[19];

  char* ws = (char*)d_ws;
  const size_t hn = (size_t)BB * NN * HID;
  float* h0 = (float*)ws;
  float* h1 = (float*)(ws + hn * 4);
  __bf16* hm0 = (__bf16*)(ws + hn * 8);
  __bf16* hm1 = (__bf16*)(ws + hn * 10);
  char* wp = ws + hn * 12;
  __bf16* wt_e2 = (__bf16*)wp;  wp += 96 * 96 * 2;
  __bf16* wt_m1[NL]; __bf16* wt_m2[NL]; __bf16* wt_u1[NL]; __bf16* wt_u2[NL];
  for (int l = 0; l < NL; ++l) {
    wt_m1[l] = (__bf16*)wp; wp += 192 * 96 * 2;
    wt_m2[l] = (__bf16*)wp; wp += 96 * 96 * 2;
    wt_u1[l] = (__bf16*)wp; wp += 96 * 192 * 2;
    wt_u2[l] = (__bf16*)wp; wp += 96 * 96 * 2;
  }
  __bf16* wt_d1 = (__bf16*)wp;  wp += 96 * 96 * 2;

  // --- weight prep (bf16 W^T layouts) ---
  transpose_bf16<<<(96 * 96 + 255) / 256, 256, 0, stream>>>(embed_w2, wt_e2, 96, 96);
  for (int l = 0; l < NL; ++l) {
    prep_msg1<<<(192 * 96 + 255) / 256, 256, 0, stream>>>(msg_w1 + (size_t)l * 193 * 96,
                                                          wt_m1[l]);
    transpose_bf16<<<(96 * 96 + 255) / 256, 256, 0, stream>>>(
        msg_w2 + (size_t)l * 96 * 96, wt_m2[l], 96, 96);
    transpose_bf16<<<(96 * 192 + 255) / 256, 256, 0, stream>>>(
        upd_w1 + (size_t)l * 192 * 96, wt_u1[l], 192, 96);
    transpose_bf16<<<(96 * 96 + 255) / 256, 256, 0, stream>>>(
        upd_w2 + (size_t)l * 96 * 96, wt_u2[l], 96, 96);
  }
  transpose_bf16<<<(96 * 96 + 255) / 256, 256, 0, stream>>>(dec_w1, wt_d1, 96, 96);

  // --- embed ---
  embed_kernel<<<dim3(NN / 128, BB), 256, 0, stream>>>(x, embed_w1, embed_b1, wt_e2,
                                                       embed_b2, h0, hm0);

  // --- layers (ping-pong h0/hm0 <-> h1/hm1) ---
  const size_t lsm = 113984 + 64 * 96 * sizeof(float);  // 138560 B dynamic LDS
  layer_kernel<<<dim3(NN / TILE, BB), 256, lsm, stream>>>(
      h0, hm0, h1, hm1, coord, msg_w1, wt_m1[0], msg_b1, wt_m2[0], msg_b2, wt_u1[0],
      upd_b1, wt_u2[0], upd_b2, ln_g, ln_b);
  layer_kernel<<<dim3(NN / TILE, BB), 256, lsm, stream>>>(
      h1, hm1, h0, hm0, coord, msg_w1 + (size_t)193 * 96, wt_m1[1], msg_b1 + 96,
      wt_m2[1], msg_b2 + 96, wt_u1[1], upd_b1 + 96, wt_u2[1], upd_b2 + 96, ln_g + 96,
      ln_b + 96);

  // --- decode ---
  decode_kernel<<<dim3(NN / 128, BB), 256, 0, stream>>>(hm0, wt_d1, dec_b1, dec_w2,
                                                        dec_b2, (float*)d_out);
}
